// U_Paddle_EMBT_68874095559199
// MI455X (gfx1250) — matrix operand order repaired and checked by reading, not compiled, not run on hardware
//
#include <hip/hip_runtime.h>

#define N_TASK   2000
#define N_SUP    25
#define N_QUERY  75
#define DIM      640
#define N_WAYS   5
#define N_LAYERS 20
#define EPSV     1e-6f

typedef __attribute__((ext_vector_type(16))) __bf16 v16bf;
typedef __attribute__((ext_vector_type(8)))  float  v8f;
typedef __attribute__((ext_vector_type(4)))  float  f4v;

union AB32 { v16bf v; f4v f[2]; };

// LDS layout (element strides chosen for bank spread + 16B alignment)
#define ZQ_S   648   // bf16 row stride, 80 rows (rows 75..79 zero)
#define ZQT_S  104   // bf16 row stride, 640 rows (cols 75..95 zero)
#define W_S    648   // bf16 row stride, 16 rows (rows 5..15 zero)
#define UCM_S  104   // bf16 row stride, 16 rows (rows 5..15, cols>=75 zero)

#define OFF_ZQ    0
#define OFF_ZQT   (80 * ZQ_S * 2)                 // 103680
#define OFF_W     (OFF_ZQT + 640 * ZQT_S * 2)     // 236800
#define OFF_UCM   (OFF_W + 16 * W_S * 2)          // 257536
#define OFF_LBUF  (OFF_UCM + 16 * UCM_S * 2)      // 260864
#define OFF_SUM   (OFF_LBUF + 80 * 16 * 4)        // 265984
#define OFF_MISC  (OFF_SUM + N_WAYS * DIM * 4)    // 278784
#define SMEM_BYTES (OFF_MISC + 160)               // 278944 < 320KB WGP LDS

__device__ __forceinline__ float softplus_f(float x) {
  return (x > 20.f) ? x : log1pf(__expf(x));
}

__device__ __forceinline__ __bf16 f2bf(float f) {
  unsigned u = __float_as_uint(f);
  unsigned r = (u + 0x7FFFu + ((u >> 16) & 1u)) >> 16;   // RNE truncation
  unsigned short h = (unsigned short)r;
  __bf16 b;
  __builtin_memcpy(&b, &h, 2);
  return b;
}

extern "C" __global__ __launch_bounds__(256)
void paddle_wmma_kernel(const float* __restrict__ gZs, const int* __restrict__ gY,
                        const float* __restrict__ gZq, const float* __restrict__ pt1,
                        const float* __restrict__ pgam, const float* __restrict__ pt2,
                        float* __restrict__ gOut) {
  extern __shared__ __align__(16) char smem[];
  __bf16* ZQ   = (__bf16*)(smem + OFF_ZQ);
  __bf16* ZQT  = (__bf16*)(smem + OFF_ZQT);
  __bf16* Wm   = (__bf16*)(smem + OFF_W);
  __bf16* UCM  = (__bf16*)(smem + OFF_UCM);
  float*  LBUF = (float*)(smem + OFF_LBUF);
  float*  SUMS = (float*)(smem + OFF_SUM);
  float*  MISC = (float*)(smem + OFF_MISC);
  float* USUM = MISC + 0;    // [5]
  float* Vv   = MISC + 8;    // [5]
  float* DEN  = MISC + 16;   // [5]
  float* WSQ  = MISC + 24;   // [5]  0.5*||W_k||^2
  float* CNT  = MISC + 32;   // [5]

  const int tid  = threadIdx.x;
  const int wave = tid >> 5;
  const int lane = tid & 31;
  const int ln   = lane & 15;
  const int hi   = lane >> 4;
  const int task = blockIdx.x;

  const float scale  = softplus_f(pt1[0]);
  const float g      = softplus_f(pgam[0]);
  const float inv_t2 = 1.f / (softplus_f(pt2[0]) + 1.f);

  const float* Zs = gZs + (size_t)task * N_SUP * DIM;
  const int*   Ys = gY  + (size_t)task * N_SUP;
  const float* Zq = gZq + (size_t)task * N_QUERY * DIM;

  // ---- zero all LDS (pads must be 0 for WMMA tiles) ----
  {
    unsigned* z = (unsigned*)smem;
    for (int i = tid; i < SMEM_BYTES / 4; i += 256) z[i] = 0u;
  }
  __syncthreads();

  // ---- load scaled Zq into ZQ (row-major) and ZQT (transposed) ----
  for (int i = tid; i < (N_QUERY * DIM) / 4; i += 256) {
    f4v v = ((const f4v*)Zq)[i];
    int q = (i * 4) / DIM;
    int d = (i * 4) % DIM;
    float f0 = v.x * scale, f1 = v.y * scale, f2 = v.z * scale, f3 = v.w * scale;
    __bf16* zr = ZQ + q * ZQ_S + d;
    zr[0] = f2bf(f0); zr[1] = f2bf(f1); zr[2] = f2bf(f2); zr[3] = f2bf(f3);
    ZQT[(d + 0) * ZQT_S + q] = f2bf(f0);
    ZQT[(d + 1) * ZQT_S + q] = f2bf(f1);
    ZQT[(d + 2) * ZQT_S + q] = f2bf(f2);
    ZQT[(d + 3) * ZQT_S + q] = f2bf(f3);
  }

  // ---- class sums of scaled support ----
  for (int d = tid; d < DIM; d += 256) {
    float a0 = 0, a1 = 0, a2 = 0, a3 = 0, a4 = 0;
    #pragma unroll 5
    for (int s = 0; s < N_SUP; ++s) {
      float z = Zs[s * DIM + d] * scale;
      int   y = Ys[s];
      a0 += (y == 0) ? z : 0.f;
      a1 += (y == 1) ? z : 0.f;
      a2 += (y == 2) ? z : 0.f;
      a3 += (y == 3) ? z : 0.f;
      a4 += (y == 4) ? z : 0.f;
    }
    SUMS[0 * DIM + d] = a0; SUMS[1 * DIM + d] = a1; SUMS[2 * DIM + d] = a2;
    SUMS[3 * DIM + d] = a3; SUMS[4 * DIM + d] = a4;
  }
  if (tid < N_WAYS) {
    int c = 0;
    for (int s = 0; s < N_SUP; ++s) c += (Ys[s] == tid) ? 1 : 0;
    CNT[tid] = (float)c;
    Vv[tid]  = 1.f;
  }
  __syncthreads();

  // ---- W0 = SUMS/CNT ; WSQ = 0.5*||W0||^2 ----
  {
    float wa[5] = {0, 0, 0, 0, 0};
    for (int d = tid; d < DIM; d += 256) {
      #pragma unroll
      for (int k = 0; k < N_WAYS; ++k) {
        float wv = SUMS[k * DIM + d] / CNT[k];
        Wm[k * W_S + d] = f2bf(wv);
        wa[k] += 0.5f * wv * wv;
      }
    }
    #pragma unroll
    for (int k = 0; k < N_WAYS; ++k) atomicAdd(&WSQ[k], wa[k]);
  }
  __syncthreads();

  // ---- iterations ----
  for (int it = 0; it < N_LAYERS; ++it) {
    if (tid < 8) USUM[tid] = 0.f;

    // matmul1: logits_raw = Zq @ W^T  -> LBUF[80][16]
    if (wave < 5) {
      const int m0 = wave * 16;
      v8f c = {0.f, 0.f, 0.f, 0.f, 0.f, 0.f, 0.f, 0.f};
      const __bf16* ar = ZQ + (m0 + ln) * ZQ_S + hi * 8;
      const __bf16* br = Wm + ln * W_S + hi * 8;
      #pragma unroll 4
      for (int ks = 0; ks < DIM / 32; ++ks) {
        AB32 a, b;
        a.f[0] = *(const f4v*)(ar);
        a.f[1] = *(const f4v*)(ar + 16);
        b.f[0] = *(const f4v*)(br);
        b.f[1] = *(const f4v*)(br + 16);
        ar += 32; br += 32;
        c = __builtin_amdgcn_wmma_f32_16x16x32_bf16(false, a.v, false, b.v,
                                                    (short)0, c, false, false);
      }
      #pragma unroll
      for (int r = 0; r < 8; ++r)
        LBUF[(m0 + r + 8 * hi) * 16 + ln] = c[r];
    }
    __syncthreads();

    // softmax over classes (q_sq is row-constant -> cancels; use row max)
    if (tid < N_QUERY) {
      const int q = tid;
      float x[N_WAYS], xmax = -1e30f;
      #pragma unroll
      for (int k = 0; k < N_WAYS; ++k) {
        x[k] = inv_t2 * (LBUF[q * 16 + k] - WSQ[k] + g * Vv[k]);
        xmax = fmaxf(xmax, x[k]);
      }
      float ssum = 0.f;
      #pragma unroll
      for (int k = 0; k < N_WAYS; ++k) { x[k] = __expf(x[k] - xmax); ssum += x[k]; }
      float inv = 1.f / ssum;
      #pragma unroll
      for (int k = 0; k < N_WAYS; ++k) {
        float u = x[k] * inv;
        UCM[k * UCM_S + q] = f2bf(u);   // A operand of matmul2
        LBUF[q * 16 + k]   = u;         // fp32 copy (becomes U_final)
        atomicAdd(&USUM[k], u);
      }
    }
    __syncthreads();

    if (tid < N_WAYS) {
      float us = USUM[tid];
      Vv[tid]  = logf(us * (1.f / N_QUERY) + EPSV) + 1.f;
      DEN[tid] = us + CNT[tid];
      WSQ[tid] = 0.f;   // re-accumulated below
    }
    __syncthreads();

    // matmul2: num = U^T @ Zq ; W = (num + SUMS) / DEN ; WSQ = 0.5||W||^2
    {
      float wa[N_WAYS] = {0, 0, 0, 0, 0};
      #pragma unroll 1
      for (int j = 0; j < 5; ++j) {
        const int d0 = (wave * 5 + j) * 16;
        const int dc = d0 + ln;
        v8f c = {0.f, 0.f, 0.f, 0.f, 0.f, 0.f, 0.f, 0.f};
        const __bf16* ar = UCM + ln * UCM_S + hi * 8;
        const __bf16* br = ZQT + dc * ZQT_S + hi * 8;
        #pragma unroll
        for (int ks = 0; ks < 3; ++ks) {       // K = 75 padded to 96
          AB32 a, b;
          a.f[0] = *(const f4v*)(ar + ks * 32);
          a.f[1] = *(const f4v*)(ar + ks * 32 + 16);
          b.f[0] = *(const f4v*)(br + ks * 32);
          b.f[1] = *(const f4v*)(br + ks * 32 + 16);
          c = __builtin_amdgcn_wmma_f32_16x16x32_bf16(false, a.v, false, b.v,
                                                      (short)0, c, false, false);
        }
        if (hi == 0) {                         // valid classes live in lanes 0..15, r<5
          #pragma unroll
          for (int r = 0; r < N_WAYS; ++r) {
            float val = (c[r] + SUMS[r * DIM + dc]) / DEN[r];
            Wm[r * W_S + dc] = f2bf(val);
            wa[r] += 0.5f * val * val;
          }
        }
      }
      if (hi == 0) {
        #pragma unroll
        for (int r = 0; r < N_WAYS; ++r) atomicAdd(&WSQ[r], wa[r]);
      }
    }
    __syncthreads();
  }

  // ---- write U_final [75][5] fp32 ----
  for (int i = tid; i < N_QUERY * N_WAYS; i += 256) {
    gOut[(size_t)task * (N_QUERY * N_WAYS) + i] = LBUF[(i / N_WAYS) * 16 + (i % N_WAYS)];
  }
}

extern "C" void kernel_launch(void* const* d_in, const int* in_sizes, int n_in,
                              void* d_out, int out_size, void* d_ws, size_t ws_size,
                              hipStream_t stream) {
  (void)in_sizes; (void)n_in; (void)d_ws; (void)ws_size; (void)out_size;
  // Allow >64KB dynamic LDS (WGP supports 320KB). Deterministic: called every time.
  hipFuncSetAttribute((const void*)paddle_wmma_kernel,
                      hipFuncAttributeMaxDynamicSharedMemorySize, SMEM_BYTES);
  dim3 grid(N_TASK), block(256);
  paddle_wmma_kernel<<<grid, block, SMEM_BYTES, stream>>>(
      (const float*)d_in[0], (const int*)d_in[1], (const float*)d_in[2],
      (const float*)d_in[3], (const float*)d_in[4], (const float*)d_in[5],
      (float*)d_out);
}
